// DecodeCell_3839700762881
// MI455X (gfx1250) — matrix operand order repaired and checked by reading, not compiled, not run on hardware
//
#include <hip/hip_runtime.h>
#include <hip/hip_bf16.h>

// ---------- types ----------
typedef __attribute__((ext_vector_type(16))) __bf16 v16bf;
typedef __attribute__((ext_vector_type(8)))  float  v8f;

#define WMMA_BF16(a, b, c) \
  __builtin_amdgcn_wmma_f32_16x16x32_bf16(false, (a), false, (b), (short)0, (c), false, false)

static __device__ __forceinline__ unsigned short f2bf(float f) {
  unsigned int u = __float_as_uint(f);
  u += 0x7fffu + ((u >> 16) & 1u);            // round-to-nearest-even
  return (unsigned short)(u >> 16);
}

// A-fragment (16x32 bf16, MxK): lane L holds row M=L%16; VGPR i holds K pair
// {(2i<8 ? 2i : 2i+8) + 8*(L>=16), +1}.  Source: LDS bf16, row stride 1024.
static __device__ __forceinline__ v16bf load_a_bf(const unsigned short* buf, int m, int kBase, int hi) {
  union { v16bf v; unsigned int u[8]; } r;
  const int base = m * 1024 + kBase + 8 * hi;
#pragma unroll
  for (int i = 0; i < 8; ++i) {
    int e = 2 * i;
    int k = (e < 8) ? e : e + 8;
    r.u[i] = *(const unsigned int*)(buf + base + k);
  }
  return r.v;
}

// Same fragment but gathered from f32 LDS (convert on the fly).
static __device__ __forceinline__ v16bf load_a_f32(const float* buf, int m, int kBase, int hi) {
  union { v16bf v; unsigned short h[16]; } r;
  const int base = m * 1024 + kBase + 8 * hi;
#pragma unroll
  for (int i = 0; i < 8; ++i) {
    int e = 2 * i;
    int k = (e < 8) ? e : e + 8;
    float2 f = *(const float2*)(buf + base + k);
    r.h[2 * i]     = f2bf(f.x);
    r.h[2 * i + 1] = f2bf(f.y);
  }
  return r.v;
}

// B-fragment: packed weights, 512 bf16 per (32x16) tile, lane-contiguous 32B.
static __device__ __forceinline__ v16bf load_b(const unsigned short* p, int tile, int lane) {
  union { v16bf v; uint4 q[2]; } r;
  const uint4* src = (const uint4*)(p + (size_t)tile * 512) + lane * 2;
  r.q[0] = src[0];
  r.q[1] = src[1];
  return r.v;
}

static __device__ __forceinline__ float hsig(float x) {
  return fminf(fmaxf(0.2f * x + 0.5f, 0.0f), 1.0f);
}

// ---------- weight packing ----------
// B layout inside a 32x16 (KxN) tile: lane L, element e -> K = (e<8 ? e : e+8) + 8*(L>=16), N = L%16.
// Tile order: tileIdx = nt*KT + kt  (K-streaming for a fixed column tile).

__global__ void pack_wv(const float* __restrict__ W, const float* __restrict__ Vr,
                        const float* __restrict__ Vz, const float* __restrict__ Vs,
                        unsigned short* __restrict__ dst) {
  int idx = blockIdx.x * blockDim.x + threadIdx.x;
  const int total = 1024 * 3072;
  if (idx >= total) return;
  const int KT = 32;
  int tile = idx >> 9, rem = idx & 511;
  int lane = rem >> 4, e = rem & 15;
  int kt = tile % KT, nt = tile / KT;
  int hi = lane >> 4;
  int n = nt * 16 + (lane & 15);
  int k = kt * 32 + ((e < 8) ? e : e + 8) + 8 * hi;
  float v;
  if (k < 512) {
    v = W[k * 3072 + n];                       // W: (512, 3072)
  } else {
    int kk = k - 512;
    if (n < 1024)      v = Vr[kk * 1024 + n];
    else if (n < 2048) v = Vz[kk * 1024 + (n - 1024)];
    else               v = Vs[kk * 1024 + (n - 2048)];
  }
  dst[idx] = f2bf(v);
}

__global__ void pack_generic(const float* __restrict__ src, int ld, int co,
                             int K, int N, unsigned short* __restrict__ dst) {
  int idx = blockIdx.x * blockDim.x + threadIdx.x;
  int total = K * N;
  if (idx >= total) return;
  int KT = K >> 5;
  int tile = idx >> 9, rem = idx & 511;
  int lane = rem >> 4, e = rem & 15;
  int kt = tile % KT, nt = tile / KT;
  int hi = lane >> 4;
  int n = nt * 16 + (lane & 15);
  int k = kt * 32 + ((e < 8) ? e : e + 8) + 8 * hi;
  dst[idx] = f2bf(src[k * ld + co + n]);
}

// ---------- persistent recurrence kernel ----------
// 8 blocks x 512 threads (16 waves).  Block owns 16 batch rows; runs all 256 steps.
// Wave w owns HID column-tiles {w, w+16, w+32, w+48} for r, z, and s gates
// (same columns for all three => z and xs accumulators stay in registers).

__global__ __launch_bounds__(512) void cell_kernel(
    const float* __restrict__ x,  const float* __restrict__ x0, const float* __restrict__ s0,
    const float* __restrict__ b,  const float* __restrict__ bx, const float* __restrict__ by,
    const unsigned short* __restrict__ pWV,  const unsigned short* __restrict__ pUrz,
    const unsigned short* __restrict__ pUs,  const unsigned short* __restrict__ pWx,
    const unsigned short* __restrict__ pWy,  float* __restrict__ out) {
  __shared__ unsigned short abuf[16 * 1024];                        // [x_prev bf16 | inp_t bf16]
  __shared__ float          sbuf[16 * 1024];                        // s state, f32
  __shared__ union RS { unsigned short rs[16 * 1024]; float o[16 * 512]; } rso;
  __shared__ float          tbuf[16 * 64];

  const int tid  = threadIdx.x;
  const int w    = tid >> 5;
  const int lane = tid & 31;
  const int m    = lane & 15;
  const int hi   = lane >> 4;
  const int hi8  = hi * 8;
  const int m0   = blockIdx.x * 16;

  // init: x_prev = x0, s = s0
  for (int i = tid; i < 16 * 512; i += 512) {
    int mm = i >> 9, d = i & 511;
    abuf[mm * 1024 + d] = f2bf(x0[(m0 + mm) * 512 + d]);
  }
  for (int i = tid; i < 16 * 1024; i += 512) {
    int mm = i >> 10, d = i & 1023;
    sbuf[i] = s0[(m0 + mm) * 1024 + d];
  }
  __syncthreads();

  const v8f zero = {0.f, 0.f, 0.f, 0.f, 0.f, 0.f, 0.f, 0.f};

  for (int t = 0; t < 256; ++t) {
    // ---- stage inp_t (f32 global -> bf16 LDS, second half of abuf) ----
    for (int i = tid; i < 16 * 512; i += 512) {
      int mm = i >> 9, d = i & 511;
      abuf[mm * 1024 + 512 + d] = f2bf(x[((size_t)(m0 + mm) * 256 + t) * 512 + d]);
    }
    __syncthreads();

    v8f accR[4], accZ[4], accS[4];
#pragma unroll
    for (int j = 0; j < 4; ++j) { accR[j] = zero; accZ[j] = zero; accS[j] = zero; }

    // ---- GEMM1: [x_prev|inp] (16x1024) @ [W;V] (1024x3072) ----
    for (int kt = 0; kt < 32; ++kt) {
      v16bf a = load_a_bf(abuf, m, kt * 32, hi);
#pragma unroll
      for (int j = 0; j < 4; ++j) {
        int ct = w + 16 * j;
        accR[j] = WMMA_BF16(a, load_b(pWV, (ct)       * 32 + kt, lane), accR[j]);
        accZ[j] = WMMA_BF16(a, load_b(pWV, (64 + ct)  * 32 + kt, lane), accZ[j]);
        accS[j] = WMMA_BF16(a, load_b(pWV, (128 + ct) * 32 + kt, lane), accS[j]);
      }
    }
    // ---- GEMM2: s (16x1024) @ U[:, :2048]  (r and z gates only) ----
    for (int kt = 0; kt < 32; ++kt) {
      v16bf a = load_a_f32(sbuf, m, kt * 32, hi);
#pragma unroll
      for (int j = 0; j < 4; ++j) {
        int ct = w + 16 * j;
        accR[j] = WMMA_BF16(a, load_b(pUrz, (ct)      * 32 + kt, lane), accR[j]);
        accZ[j] = WMMA_BF16(a, load_b(pUrz, (64 + ct) * 32 + kt, lane), accZ[j]);
      }
    }
    // ---- elementwise: r = hsig(.), write r*s (bf16), z kept in regs ----
#pragma unroll
    for (int j = 0; j < 4; ++j) {
      int ct = w + 16 * j;
      int n  = ct * 16 + (lane & 15);
      float br = b[n], bz = b[1024 + n];
#pragma unroll
      for (int i = 0; i < 8; ++i) {
        int mm = i + hi8;
        float r    = hsig(accR[j][i] + br);
        float sold = sbuf[mm * 1024 + n];
        rso.rs[mm * 1024 + n] = f2bf(r * sold);
        accZ[j][i] = hsig(accZ[j][i] + bz);
      }
    }
    __syncthreads();

    // ---- GEMM3: (r*s) (16x1024) @ U_s (1024x1024), acc seeded with xs ----
    for (int kt = 0; kt < 32; ++kt) {
      v16bf a = load_a_bf(rso.rs, m, kt * 32, hi);
#pragma unroll
      for (int j = 0; j < 4; ++j) {
        int ct = w + 16 * j;
        accS[j] = WMMA_BF16(a, load_b(pUs, ct * 32 + kt, lane), accS[j]);
      }
    }
    // ---- state update: s_t = (1-z)s + z*tanh(xs + (r*s)Us + b_s) ----
#pragma unroll
    for (int j = 0; j < 4; ++j) {
      int ct = w + 16 * j;
      int n  = ct * 16 + (lane & 15);
      float bs = b[2048 + n];
#pragma unroll
      for (int i = 0; i < 8; ++i) {
        int mm = i + hi8;
        float s1   = tanhf(accS[j][i] + bs);
        float z    = accZ[j][i];
        float sold = sbuf[mm * 1024 + n];
        sbuf[mm * 1024 + n] = (1.0f - z) * sold + z * s1;
      }
    }
    __syncthreads();

    // ---- GEMM4: s_t (16x1024) @ W_x (1024x512) -> logits ----
    v8f accO[2] = {zero, zero};
    for (int kt = 0; kt < 32; ++kt) {
      v16bf a = load_a_f32(sbuf, m, kt * 32, hi);
#pragma unroll
      for (int j = 0; j < 2; ++j) {
        int nt = w + 16 * j;
        accO[j] = WMMA_BF16(a, load_b(pWx, nt * 32 + kt, lane), accO[j]);
      }
    }
#pragma unroll
    for (int j = 0; j < 2; ++j) {
      int nt = w + 16 * j;
      int n  = nt * 16 + (lane & 15);
      float bxv = bx[n];
#pragma unroll
      for (int i = 0; i < 8; ++i) rso.o[(i + hi8) * 512 + n] = accO[j][i] + bxv;
    }
    __syncthreads();

    // ---- softmax over 512 cols, one row per wave; probs -> abuf (= next x_prev) ----
    {
      int row = w;
      float vals[16];
      float mx = -3.4e38f;
#pragma unroll
      for (int i = 0; i < 16; ++i) {
        float v = rso.o[row * 512 + lane + 32 * i];
        vals[i] = v;
        mx = fmaxf(mx, v);
      }
      for (int off = 16; off; off >>= 1) mx = fmaxf(mx, __shfl_xor(mx, off, 32));
      float sum = 0.f;
#pragma unroll
      for (int i = 0; i < 16; ++i) { vals[i] = __expf(vals[i] - mx); sum += vals[i]; }
      for (int off = 16; off; off >>= 1) sum += __shfl_xor(sum, off, 32);
      float inv = 1.0f / sum;
#pragma unroll
      for (int i = 0; i < 16; ++i) abuf[row * 1024 + lane + 32 * i] = f2bf(vals[i] * inv);
    }
    __syncthreads();

    // ---- GEMM5: x_t (16x512) @ W_y (512x64) ----
    if (w < 4) {
      v8f accT = zero;
      for (int kt = 0; kt < 16; ++kt) {
        v16bf a = load_a_bf(abuf, m, kt * 32, hi);
        accT = WMMA_BF16(a, load_b(pWy, w * 16 + kt, lane), accT);
      }
      int n = w * 16 + (lane & 15);
      float byv = by[n];
#pragma unroll
      for (int i = 0; i < 8; ++i) tbuf[(i + hi8) * 64 + n] = accT[i] + byv;
    }
    __syncthreads();

    // ---- softmax over 64 cols, one row per wave; write output f32 ----
    {
      int row = w;
      float v0 = tbuf[row * 64 + lane];
      float v1 = tbuf[row * 64 + lane + 32];
      float mx = fmaxf(v0, v1);
      for (int off = 16; off; off >>= 1) mx = fmaxf(mx, __shfl_xor(mx, off, 32));
      float e0 = __expf(v0 - mx), e1 = __expf(v1 - mx);
      float s = e0 + e1;
      for (int off = 16; off; off >>= 1) s += __shfl_xor(s, off, 32);
      float inv = 1.0f / s;
      float* o = out + ((size_t)(m0 + row) * 256 + t) * 64;
      o[lane]      = e0 * inv;
      o[lane + 32] = e1 * inv;
    }
    __syncthreads();
  }
}

extern "C" void kernel_launch(void* const* d_in, const int* in_sizes, int n_in,
                              void* d_out, int out_size, void* d_ws, size_t ws_size,
                              hipStream_t stream) {
  const float* x  = (const float*)d_in[0];
  const float* x0 = (const float*)d_in[1];
  const float* s0 = (const float*)d_in[2];
  const float* W  = (const float*)d_in[3];
  const float* U  = (const float*)d_in[4];
  const float* b  = (const float*)d_in[5];
  const float* Wx = (const float*)d_in[6];
  const float* bx = (const float*)d_in[7];
  const float* Vr = (const float*)d_in[8];
  const float* Vz = (const float*)d_in[9];
  const float* Vs = (const float*)d_in[10];
  const float* Wy = (const float*)d_in[11];
  const float* by = (const float*)d_in[12];

  unsigned short* ws   = (unsigned short*)d_ws;
  unsigned short* pWV  = ws;                         // 1024*3072
  unsigned short* pUrz = pWV  + 1024 * 3072;         // 1024*2048
  unsigned short* pUs  = pUrz + 1024 * 2048;         // 1024*1024
  unsigned short* pWx  = pUs  + 1024 * 1024;         // 1024*512
  unsigned short* pWy  = pWx  + 1024 * 512;          // 512*64

  pack_wv<<<(1024 * 3072 + 255) / 256, 256, 0, stream>>>(W, Vr, Vz, Vs, pWV);
  pack_generic<<<(1024 * 2048 + 255) / 256, 256, 0, stream>>>(U, 3072, 0, 1024, 2048, pUrz);
  pack_generic<<<(1024 * 1024 + 255) / 256, 256, 0, stream>>>(U, 3072, 2048, 1024, 1024, pUs);
  pack_generic<<<(1024 * 512 + 255) / 256, 256, 0, stream>>>(Wx, 512, 0, 1024, 512, pWx);
  pack_generic<<<(512 * 64 + 255) / 256, 256, 0, stream>>>(Wy, 64, 0, 512, 64, pWy);

  cell_kernel<<<8, 512, 0, stream>>>(x, x0, s0, b, bx, by,
                                     pWV, pUrz, pUs, pWx, pWy, (float*)d_out);
}
